// Model_82566451298337
// MI455X (gfx1250) — compile-verified, matrix operand order checked by reading, not run on hardware
//
#include <hip/hip_runtime.h>
#include <stdint.h>

// Binarized CNN. All convs are (+/-1)x(+/-1) integer dot products:
//  - layer 1 (K=81): XNOR-popcount in 3x27-bit words (lane = out-channel)
//  - layers 2,3 (K=2592/1296): V_WMMA_I32_16X16X64_IU8, signed A x signed B
//  - layer 4 (K=128): trivial scalar
// BN(affine=False)+sign == (pre >= ceil(sum/NHW)) exactly, so binarization is a
// per-channel integer threshold fused into the consumer's LDS-fill (no extra
// global passes). All accumulation is integer atomics -> deterministic.

typedef __attribute__((ext_vector_type(8))) int v8i;

#define NIMG 8192

// Workspace layout (bytes)
#define OFF_SUMS1   0          // 32 x i64
#define OFF_SUMS2   256        // 16 x i64
#define OFF_SUMS3   512        //  8 x i64
#define OFF_PACKB2  1024       // 41*32*8 u32 = 41984 B (layer2 B fragments)
#define OFF_PACKB3  43264      // 21*32*8 u32 = 21504 B (layer3 B fragments)
#define OFF_PRE1    65536      // int8 NHWC (8192,20,20,32) = 104857600 B
#define OFF_PRE2    104923136  // int16 (8192,144,16) = 37748736 B
#define OFF_PRE3    142671872  // int16 (8192,16,8) = 2097152 B

__device__ __forceinline__ int kbyte_off(int v, int hi) {
    // IU8 fragment byte offset within a 64-wide K chunk (ISA 7.12.2, 8-bit 16x64)
    return ((v >> 1) << 4) + ((v & 1) << 2) + (hi << 3);
}
__device__ __forceinline__ signed char sgn8f(float f) {
    return f >= 0.f ? (signed char)1 : (signed char)-1;
}
// threshold t s.t. (pre*NHW - sum >= 0) <=> (pre >= t), t = ceil(sum/NHW)
__device__ __forceinline__ int ceil_div_ll(long long s, long long nhw) {
    long long q = s / nhw, r = s % nhw;
    return (int)(q + (r > 0 ? 1 : 0));
}
__device__ __forceinline__ unsigned bin4(unsigned w, int t0, int t1, int t2, int t3) {
    unsigned o;
    o  = (unsigned)(unsigned char)((((int)(signed char)(w      )) >= t0) ? 1 : -1);
    o |= (unsigned)(unsigned char)((((int)(signed char)(w >>  8)) >= t1) ? 1 : -1) << 8;
    o |= (unsigned)(unsigned char)((((int)(signed char)(w >> 16)) >= t2) ? 1 : -1) << 16;
    o |= (unsigned)(unsigned char)((((int)w >> 24)                >= t3) ? 1 : -1) << 24;
    return o;
}
__device__ __forceinline__ unsigned bin2s(unsigned w, int tlo, int thi) {
    int lo = (int)(short)(w & 0xFFFFu);
    int hi = (int)w >> 16;
    return (unsigned)(unsigned char)((lo >= tlo) ? 1 : -1)
         | ((unsigned)(unsigned char)((hi >= thi) ? 1 : -1) << 8);
}

// ---------------- zero sum accumulators ----------------
__global__ void k_zero(unsigned long long* s) {
    if (threadIdx.x < 128) s[threadIdx.x] = 0ull;
}

// ---------------- pack +/-1 weights into IU8 B-fragment order ----------------
// Layer2: K order = pos*32+ic (pos 0..80, ic 0..31), 41 chunks of 64.
// Layer3: K order = pos*16+ic (ic 0..15), 21 chunks; N padded 8->16 with zeros.
__global__ void k_pack(const float* __restrict__ w2, const float* __restrict__ w3,
                       unsigned* __restrict__ pb2, unsigned* __restrict__ pb3) {
    const int total2 = 41 * 32 * 8, total3 = 21 * 32 * 8;
    for (int i = blockIdx.x * blockDim.x + threadIdx.x; i < total2 + total3;
         i += gridDim.x * blockDim.x) {
        if (i < total2) {
            int v = i & 7, lane = (i >> 3) & 31, c = i >> 8;
            int ko = kbyte_off(v, lane >> 4), n = lane & 15;
            unsigned word = 0;
            for (int j = 0; j < 4; ++j) {
                int kg = c * 64 + ko + j;
                int pos = kg >> 5, ic = kg & 31;
                signed char b = 0;
                if (pos < 81) b = sgn8f(w2[(n * 32 + ic) * 81 + pos]);
                word |= ((unsigned)(unsigned char)b) << (8 * j);
            }
            pb2[i] = word;
        } else {
            int i3 = i - total2;
            int v = i3 & 7, lane = (i3 >> 3) & 31, c = i3 >> 8;
            int ko = kbyte_off(v, lane >> 4), n = lane & 15;
            unsigned word = 0;
            for (int j = 0; j < 4; ++j) {
                int kg = c * 64 + ko + j;
                int pos = kg >> 4, ic = kg & 15;
                signed char b = 0;
                if (pos < 81 && n < 8) b = sgn8f(w3[(n * 16 + ic) * 81 + pos]);
                word |= ((unsigned)(unsigned char)b) << (8 * j);
            }
            pb3[i3] = word;
        }
    }
}

// ---------------- layer 1: 1->32ch 9x9, XNOR-popcount ----------------
// lane = out-channel (per-lane sum accumulator), wave handles pixels.
// Patch/filter packed as 3 words of 27 bits (3 kernel rows x 9 taps each).
#define IMGS1 16
__global__ void k_conv1(const float* __restrict__ x, const float* __restrict__ w1,
                        const float* __restrict__ b1, signed char* __restrict__ pre1,
                        unsigned long long* __restrict__ sums1) {
    __shared__ unsigned xrow[28];   // bit i = (x[row][i] >= 0)
    __shared__ unsigned wb[32][3];
    __shared__ int sbb[32];
    __shared__ int csum[32];
    const int tid = threadIdx.x, wv = tid >> 5, lane = tid & 31;
    if (tid >= 32 && tid < 128) {
        int idx = tid - 32, oc = idx / 3, j = idx % 3;
        const float* wp = w1 + oc * 81 + j * 27;
        unsigned m = 0;
        for (int k = 0; k < 27; ++k) m |= (wp[k] >= 0.f ? 1u : 0u) << k;
        wb[oc][j] = m;
    }
    if (tid < 32) { sbb[tid] = b1[tid] >= 0.f ? 1 : -1; csum[tid] = 0; }
    __syncthreads();
    const unsigned f0 = wb[lane][0], f1 = wb[lane][1], f2 = wb[lane][2];
    const int sb = sbb[lane];
    int acc = 0;
    for (int it = 0; it < IMGS1; ++it) {
        const int img = blockIdx.x * IMGS1 + it;
        __syncthreads();                       // xrow reuse hazard
        if (tid < 28) {
            const float* xr = x + (size_t)img * 784 + tid * 28;
            unsigned m = 0;
            for (int i = 0; i < 28; ++i) m |= (xr[i] >= 0.f ? 1u : 0u) << i;
            xrow[tid] = m;
        }
        __syncthreads();
        signed char* op = pre1 + (size_t)img * 12800;
        for (int p = wv; p < 400; p += 8) {
            int oy = p / 20, ox = p % 20;
            unsigned a0 = ((xrow[oy    ] >> ox) & 0x1FF)
                        | (((xrow[oy + 1] >> ox) & 0x1FF) << 9)
                        | (((xrow[oy + 2] >> ox) & 0x1FF) << 18);
            unsigned a1 = ((xrow[oy + 3] >> ox) & 0x1FF)
                        | (((xrow[oy + 4] >> ox) & 0x1FF) << 9)
                        | (((xrow[oy + 5] >> ox) & 0x1FF) << 18);
            unsigned a2 = ((xrow[oy + 6] >> ox) & 0x1FF)
                        | (((xrow[oy + 7] >> ox) & 0x1FF) << 9)
                        | (((xrow[oy + 8] >> ox) & 0x1FF) << 18);
            int mism = __popc(a0 ^ f0) + __popc(a1 ^ f1) + __popc(a2 ^ f2);
            int val = 81 - 2 * mism + sb;      // |val| <= 82
            op[p * 32 + lane] = (signed char)val;
            acc += val;
        }
    }
    atomicAdd(&csum[lane], acc);
    __syncthreads();
    if (tid < 32) atomicAdd(&sums1[tid], (unsigned long long)(long long)csum[tid]);
}

// ---------------- layer 2: 32->16ch 9x9, IU8 WMMA (dominant) ----------------
// Block = 9 waves, 4 images serially. B fragments + binarized image in LDS.
#define IMGS2 4
__global__ void k_conv2(const signed char* __restrict__ pre1,
                        const unsigned* __restrict__ pb2, const float* __restrict__ b2,
                        const unsigned long long* __restrict__ sums1,
                        short* __restrict__ pre2, unsigned long long* __restrict__ sums2) {
    __shared__ __align__(16) unsigned simg[3208];  // 20*20*32 B + 32 B zero pad
    __shared__ v8i sB[41 * 32];                     // 41984 B packed B fragments
    __shared__ int thr[32];
    __shared__ unsigned long long bsum[16];
    const int tid = threadIdx.x;
    {   // stage B fragments
        const v8i* src = (const v8i*)pb2;
        for (int i = tid; i < 41 * 32; i += 288) sB[i] = src[i];
    }
    if (tid < 32) thr[tid] = ceil_div_ll((long long)sums1[tid], 8192LL * 400LL);
    if (tid < 16) bsum[tid] = 0;
    for (int i = 3200 + tid; i < 3208; i += 288) simg[i] = 0;  // K-tail zero pad
    __syncthreads();
    // per-thread threshold regs: uint4 element e, byte j -> channel 16*(i&1)+4e+j
    const int i1 = tid & 1;
    int tt[16];
#pragma unroll
    for (int k = 0; k < 16; ++k) tt[k] = thr[16 * i1 + k];

    const int wv = tid >> 5, lane = tid & 31;
    const int hi = lane >> 4, n = lane & 15;
    const int pA = wv * 16 + (lane & 15);
    const int laneOff = ((pA / 12) * 20 + (pA % 12)) * 32;
    const int ic0 = hi * 8, ic1 = 4 + hi * 8, ic2 = 16 + hi * 8, ic3 = 20 + hi * 8;
    const int sbn = b2[n] >= 0.f ? 1 : -1;
    const signed char* s8 = (const signed char*)simg;
    long long lsum = 0;

    for (int it = 0; it < IMGS2; ++it) {
        const int img = blockIdx.x * IMGS2 + it;
        __syncthreads();                       // previous image fully consumed
        {   // fill LDS with binarized act1 (fused BN+sign)
            const uint4* g4 = (const uint4*)(pre1 + (size_t)img * 12800);
            uint4* s4 = (uint4*)simg;
            for (int i = tid; i < 800; i += 288) {
                uint4 d = g4[i];
                uint4 o;
                o.x = bin4(d.x, tt[0],  tt[1],  tt[2],  tt[3]);
                o.y = bin4(d.y, tt[4],  tt[5],  tt[6],  tt[7]);
                o.z = bin4(d.z, tt[8],  tt[9],  tt[10], tt[11]);
                o.w = bin4(d.w, tt[12], tt[13], tt[14], tt[15]);
                s4[i] = o;
            }
        }
        __syncthreads();
        v8i C = {};
        for (int c = 0; c < 40; ++c) {         // positions 2c, 2c+1 (all < 81)
            int posA = 2 * c, posB = 2 * c + 1;
            int dA = laneOff + ((posA / 9) * 20 + (posA % 9)) * 32;
            int dB = laneOff + ((posB / 9) * 20 + (posB % 9)) * 32;
            v8i A;
            A[0] = *(const int*)(s8 + dA + ic0);
            A[1] = *(const int*)(s8 + dA + ic1);
            A[2] = *(const int*)(s8 + dA + ic2);
            A[3] = *(const int*)(s8 + dA + ic3);
            A[4] = *(const int*)(s8 + dB + ic0);
            A[5] = *(const int*)(s8 + dB + ic1);
            A[6] = *(const int*)(s8 + dB + ic2);
            A[7] = *(const int*)(s8 + dB + ic3);
            C = __builtin_amdgcn_wmma_i32_16x16x64_iu8(true, A, true, sB[c * 32 + lane],
                                                       C, false, false);
        }
        {   // tail chunk 40: pos 80 in low half, zero pad in high half
            int dT = laneOff + (8 * 20 + 8) * 32;
            v8i A;
            A[0] = *(const int*)(s8 + dT + ic0);
            A[1] = *(const int*)(s8 + dT + ic1);
            A[2] = *(const int*)(s8 + dT + ic2);
            A[3] = *(const int*)(s8 + dT + ic3);
            A[4] = *(const int*)(s8 + 12800 + ic0);
            A[5] = *(const int*)(s8 + 12800 + ic1);
            A[6] = *(const int*)(s8 + 12800 + ic2);
            A[7] = *(const int*)(s8 + 12800 + ic3);
            C = __builtin_amdgcn_wmma_i32_16x16x64_iu8(true, A, true, sB[40 * 32 + lane],
                                                       C, false, false);
        }
#pragma unroll
        for (int g8 = 0; g8 < 8; ++g8) {
            int p = wv * 16 + g8 + 8 * hi;
            int val = C[g8] + sbn;             // |val| <= 2593
            pre2[((size_t)img * 144 + p) * 16 + n] = (short)val;
            lsum += val;
        }
    }
    atomicAdd(&bsum[n], (unsigned long long)lsum);
    __syncthreads();
    if (tid < 16) atomicAdd(&sums2[tid], bsum[tid]);
}

// ---------------- layer 3: 16->8ch 9x9, IU8 WMMA ----------------
// 8 waves/block, 1 image/wave; M = all 16 output pixels, N padded 8->16.
__global__ void k_conv3(const short* __restrict__ pre2,
                        const unsigned* __restrict__ pb3, const float* __restrict__ b3,
                        const unsigned long long* __restrict__ sums2,
                        short* __restrict__ pre3, unsigned long long* __restrict__ sums3) {
    __shared__ __align__(16) unsigned simg[8 * 580]; // per wave: 2304 B + 16 B pad
    __shared__ v8i sB[21 * 32];                       // 21504 B
    __shared__ int thr[16];
    __shared__ unsigned long long bsum[8];
    const int tid = threadIdx.x, wv = tid >> 5, lane = tid & 31;
    {
        const v8i* src = (const v8i*)pb3;
        for (int i = tid; i < 21 * 32; i += 256) sB[i] = src[i];
    }
    if (tid < 16) thr[tid] = ceil_div_ll((long long)sums2[tid], 8192LL * 144LL);
    if (tid < 8) bsum[tid] = 0;
    __syncthreads();

    const int img = blockIdx.x * 8 + wv;
    unsigned* my = &simg[wv * 580];
    {   // fill: binarize pre2 (uint2 = 4 shorts); channel = (4i+j)&15, i&3==lane&3
        int t0 = thr[4 * (lane & 3) + 0], t1 = thr[4 * (lane & 3) + 1];
        int t2 = thr[4 * (lane & 3) + 2], t3 = thr[4 * (lane & 3) + 3];
        const uint2* g2 = (const uint2*)(pre2 + (size_t)img * 2304);
        for (int i = lane; i < 576; i += 32) {
            uint2 d = g2[i];
            my[i] = bin2s(d.x, t0, t1) | (bin2s(d.y, t2, t3) << 16);
        }
        for (int i = 576 + lane; i < 580; i += 32) my[i] = 0;
    }
    __syncthreads();
    const signed char* s8 = (const signed char*)my;
    const int hi = lane >> 4, n = lane & 15, m = lane & 15;
    const int laneOff = ((m >> 2) * 12 + (m & 3)) * 16;
    const int j0 = hi * 8, j1 = 4 + hi * 8;
    v8i C = {};
    for (int c = 0; c < 20; ++c) {             // positions 4c..4c+3 (all < 81)
        v8i A;
#pragma unroll
        for (int k = 0; k < 4; ++k) {
            int pos = 4 * c + k;
            int d = laneOff + ((pos / 9) * 12 + (pos % 9)) * 16;
            A[2 * k]     = *(const int*)(s8 + d + j0);
            A[2 * k + 1] = *(const int*)(s8 + d + j1);
        }
        C = __builtin_amdgcn_wmma_i32_16x16x64_iu8(true, A, true, sB[c * 32 + lane],
                                                   C, false, false);
    }
    {   // tail chunk 20: pos 80 (v=0,1), rest zero pad
        int dT = laneOff + (8 * 12 + 8) * 16;
        v8i A;
        A[0] = *(const int*)(s8 + dT + j0);
        A[1] = *(const int*)(s8 + dT + j1);
#pragma unroll
        for (int k = 1; k < 4; ++k) {
            A[2 * k]     = *(const int*)(s8 + 2304 + j0);
            A[2 * k + 1] = *(const int*)(s8 + 2304 + j1);
        }
        C = __builtin_amdgcn_wmma_i32_16x16x64_iu8(true, A, true, sB[20 * 32 + lane],
                                                   C, false, false);
    }
    if (n < 8) {
        const int sbn = b3[n] >= 0.f ? 1 : -1;
        long long lsum = 0;
#pragma unroll
        for (int g8 = 0; g8 < 8; ++g8) {
            int p = g8 + 8 * hi;
            int val = C[g8] + sbn;             // |val| <= 1297
            pre3[((size_t)img * 16 + p) * 8 + n] = (short)val;
            lsum += val;
        }
        atomicAdd(&bsum[n], (unsigned long long)lsum);
    }
    __syncthreads();
    if (tid < 8) atomicAdd(&sums3[tid], bsum[tid]);
}

// ---------------- layer 4: 8->10ch 4x4 (K=128), fused binarize ----------------
__global__ void k_conv4(const short* __restrict__ pre3,
                        const unsigned long long* __restrict__ sums3,
                        const float* __restrict__ wl, const float* __restrict__ bl,
                        float* __restrict__ out) {
    __shared__ signed char swl[10 * 128];      // [oc][pos*8+ic] weight signs
    __shared__ int t3[8];
    const int tid = threadIdx.x;
    for (int k = tid; k < 1280; k += 256) {
        int oc = k >> 7, e = k & 127, pos = e >> 3, ic = e & 7;
        swl[k] = sgn8f(wl[(oc * 8 + ic) * 16 + pos]);
    }
    if (tid < 8) t3[tid] = ceil_div_ll((long long)sums3[tid], 8192LL * 16LL);
    __syncthreads();
    int t = blockIdx.x * blockDim.x + tid;
    if (t >= NIMG * 10) return;
    int img = t / 10, oc = t % 10;
    const short* a = pre3 + (size_t)img * 128;
    const signed char* w = &swl[oc * 128];
    int acc = bl[oc] >= 0.f ? 1 : -1;
    for (int pos = 0; pos < 16; ++pos) {
#pragma unroll
        for (int ic = 0; ic < 8; ++ic) {
            int e = pos * 8 + ic;
            int av = (a[e] >= t3[ic]) ? 1 : -1;
            acc += av * (int)w[e];
        }
    }
    out[t] = (float)acc;
}

extern "C" void kernel_launch(void* const* d_in, const int* in_sizes, int n_in,
                              void* d_out, int out_size, void* d_ws, size_t ws_size,
                              hipStream_t stream) {
    const float* x  = (const float*)d_in[0];
    const float* w1 = (const float*)d_in[1];
    const float* b1 = (const float*)d_in[2];
    const float* w2 = (const float*)d_in[3];
    const float* b2 = (const float*)d_in[4];
    const float* w3 = (const float*)d_in[5];
    const float* b3 = (const float*)d_in[6];
    const float* wl = (const float*)d_in[7];
    const float* bl = (const float*)d_in[8];

    char* ws = (char*)d_ws;
    unsigned long long* sums1 = (unsigned long long*)(ws + OFF_SUMS1);
    unsigned long long* sums2 = (unsigned long long*)(ws + OFF_SUMS2);
    unsigned long long* sums3 = (unsigned long long*)(ws + OFF_SUMS3);
    unsigned* pb2 = (unsigned*)(ws + OFF_PACKB2);
    unsigned* pb3 = (unsigned*)(ws + OFF_PACKB3);
    signed char* pre1 = (signed char*)(ws + OFF_PRE1);
    short*       pre2 = (short*)(ws + OFF_PRE2);
    short*       pre3 = (short*)(ws + OFF_PRE3);

    k_zero <<<1, 128, 0, stream>>>(sums1);
    k_pack <<<64, 256, 0, stream>>>(w2, w3, pb2, pb3);
    k_conv1<<<NIMG / IMGS1, 256, 0, stream>>>(x, w1, b1, pre1, sums1);
    k_conv2<<<NIMG / IMGS2, 288, 0, stream>>>(pre1, pb2, b2, sums1, pre2, sums2);
    k_conv3<<<NIMG / 8, 256, 0, stream>>>(pre2, pb3, b3, sums2, pre3, sums3);
    k_conv4<<<(NIMG * 10 + 255) / 256, 256, 0, stream>>>(pre3, sums3, wl, bl,
                                                         (float*)d_out);
}
